// BertSelfAttention_79448305042103
// MI455X (gfx1250) — 8-bit matrix operand order repaired and checked by reading, compiled, not run on hardware
//
#include <hip/hip_runtime.h>
#include <stdint.h>


typedef __attribute__((ext_vector_type(8))) int v8i;

#define B_   8
#define S_   512
#define H_   768
#define NH_  12
#define DH_  64
#define BS_  (B_ * S_)
#define BSH_ (BS_ * H_)
#define HH_  (H_ * H_)

__device__ __forceinline__ float wave_max_f(float v) {
#pragma unroll
  for (int off = 16; off > 0; off >>= 1)
    v = fmaxf(v, __shfl_xor(v, off, 32));
  return v;
}
__device__ __forceinline__ float wave_sum_f(float v) {
#pragma unroll
  for (int off = 16; off > 0; off >>= 1)
    v += __shfl_xor(v, off, 32);
  return v;
}
__device__ __forceinline__ unsigned long long wave_sum_u64(unsigned long long v) {
#pragma unroll
  for (int off = 16; off > 0; off >>= 1) {
    unsigned lo = (unsigned)v, hi = (unsigned)(v >> 32);
    lo = __shfl_xor(lo, off, 32);
    hi = __shfl_xor(hi, off, 32);
    v += ((unsigned long long)hi << 32) | lo;
  }
  return v;
}

__device__ __forceinline__ v8i load_a_frag(const int* __restrict__ r, int kd, int hi) {
  v8i a;
#pragma unroll
  for (int v = 0; v < 8; ++v) {
    const int base = (v >> 2) * 32, vv = v & 3;
    const int ka = (vv < 2) ? (base + hi * 8 + vv * 4)
                            : (base + 16 + hi * 8 + (vv - 2) * 4);
    a[v] = r[kd + (ka >> 2)];
  }
  return a;
}
__device__ __forceinline__ v8i load_b_frag(const int* __restrict__ r, int kd, int hi) {
  v8i b;
#pragma unroll
  for (int v = 0; v < 8; ++v) {
    const int base = (v >> 2) * 32, vv = v & 3;
    b[v] = r[kd + ((base + (vv >> 1) * 16 + hi * 8 + (vv & 1) * 4) >> 2)];
  }
  return b;
}

__global__ void k_init(unsigned long long* wsum, unsigned* maxb) {
  int t = threadIdx.x;
  if (t < 6)  wsum[t] = 0ull;
  if (t < 10) maxb[t] = 0u;
}

__global__ void k_absmax_clip(const float* __restrict__ x, int n,
                              const float* __restrict__ clip,
                              unsigned* __restrict__ out) {
  float lo = clip[0], hi = clip[1];
  float m = 0.0f;
  for (int i = blockIdx.x * blockDim.x + threadIdx.x; i < n;
       i += gridDim.x * blockDim.x) {
    float v = fminf(fmaxf(x[i], lo), hi);
    m = fmaxf(m, fabsf(v));
  }
  m = wave_max_f(m);
  if ((threadIdx.x & 31) == 0) atomicMax(out, __float_as_uint(m));
}

__global__ void k_wsign(const float* __restrict__ w, signed char* __restrict__ ws,
                        unsigned long long* __restrict__ wsum) {
  unsigned long long acc = 0ull;
  for (int i = blockIdx.x * blockDim.x + threadIdx.x; i < HH_;
       i += gridDim.x * blockDim.x) {
    float v = w[i];
    ws[i] = (signed char)((v > 0.0f) ? 1 : ((v < 0.0f) ? -1 : 0));
    acc += (unsigned long long)llrintf(fabsf(v) * 16777216.0f);
  }
  acc = wave_sum_u64(acc);
  if ((threadIdx.x & 31) == 0) atomicAdd(wsum, acc);
}

__global__ void k_scales(const unsigned long long* __restrict__ wsum,
                         const unsigned* __restrict__ maxb,
                         float* __restrict__ dqf) {
  int t = threadIdx.x;
  if (t < 6) {
    const int br = t / 3;
    const float mh = __uint_as_float(maxb[br]);
    const float alpha =
        (float)((double)wsum[t] * (1.0 / 16777216.0) / (double)HH_);
    dqf[t] = mh * alpha * (1.0f / 127.0f);
  }
}

__global__ void k_quant_act(const float* __restrict__ x, signed char* __restrict__ q,
                            int n, const float* __restrict__ clip,
                            const unsigned* __restrict__ mbits) {
  float lo = clip[0], hi = clip[1];
  float s  = 127.0f / __uint_as_float(*mbits);
  for (int i = blockIdx.x * blockDim.x + threadIdx.x; i < n;
       i += gridDim.x * blockDim.x) {
    float v = fminf(fmaxf(x[i], lo), hi);
    q[i] = (signed char)rintf(v * s);
  }
}

__global__ void __launch_bounds__(256)
k_proj_gemm(const signed char* __restrict__ X, const signed char* __restrict__ Wsg,
            int* __restrict__ Y, const float* __restrict__ dqf_p,
            const float* __restrict__ yclip, unsigned* __restrict__ my_bits) {
  const int lane = threadIdx.x & 31;
  const int wave = threadIdx.x >> 5;
  const int task = blockIdx.x * 8 + wave;
  const int mt = task / 12, ng = task - mt * 12;
  const int hi  = lane >> 4;
  const int l15 = lane & 15;

  const int* Xr  = (const int*)(X   + (mt * 16 + l15) * H_);
  const int* Wr0 = (const int*)(Wsg + (ng * 64 + l15) * H_);
  const int ws16 = 16 * H_ / 4;

  v8i acc0 = {}, acc1 = {}, acc2 = {}, acc3 = {};
  v8i a  = load_a_frag(Xr, 0, hi);
  v8i b0 = load_b_frag(Wr0, 0, hi);
  v8i b1 = load_b_frag(Wr0 + ws16, 0, hi);
  v8i b2 = load_b_frag(Wr0 + 2 * ws16, 0, hi);
  v8i b3 = load_b_frag(Wr0 + 3 * ws16, 0, hi);
  for (int kc = 64; kc < H_; kc += 64) {
    const int kd = kc >> 2;
    v8i an  = load_a_frag(Xr, kd, hi);
    v8i b0n = load_b_frag(Wr0, kd, hi);
    v8i b1n = load_b_frag(Wr0 + ws16, kd, hi);
    v8i b2n = load_b_frag(Wr0 + 2 * ws16, kd, hi);
    v8i b3n = load_b_frag(Wr0 + 3 * ws16, kd, hi);
    acc0 = __builtin_amdgcn_wmma_i32_16x16x64_iu8(true, a, true, b0, acc0, false, false);
    acc1 = __builtin_amdgcn_wmma_i32_16x16x64_iu8(true, a, true, b1, acc1, false, false);
    acc2 = __builtin_amdgcn_wmma_i32_16x16x64_iu8(true, a, true, b2, acc2, false, false);
    acc3 = __builtin_amdgcn_wmma_i32_16x16x64_iu8(true, a, true, b3, acc3, false, false);
    a = an; b0 = b0n; b1 = b1n; b2 = b2n; b3 = b3n;
  }
  acc0 = __builtin_amdgcn_wmma_i32_16x16x64_iu8(true, a, true, b0, acc0, false, false);
  acc1 = __builtin_amdgcn_wmma_i32_16x16x64_iu8(true, a, true, b1, acc1, false, false);
  acc2 = __builtin_amdgcn_wmma_i32_16x16x64_iu8(true, a, true, b2, acc2, false, false);
  acc3 = __builtin_amdgcn_wmma_i32_16x16x64_iu8(true, a, true, b3, acc3, false, false);

  const float dq = *dqf_p;
  const float lo = yclip[0], hic = yclip[1];
  float mloc = 0.0f;
#pragma unroll
  for (int j = 0; j < 4; ++j) {
    const v8i acc = (j == 0) ? acc0 : (j == 1) ? acc1 : (j == 2) ? acc2 : acc3;
    const int N = ng * 64 + j * 16 + l15;
#pragma unroll
    for (int r = 0; r < 8; ++r) {
      const int M = mt * 16 + r + hi * 8;
      const int cv = acc[r];
      Y[M * H_ + N] = cv;
      float y = fminf(fmaxf((float)cv * dq, lo), hic);
      mloc = fmaxf(mloc, fabsf(y));
    }
  }
  mloc = wave_max_f(mloc);
  if (lane == 0) atomicMax(my_bits, __float_as_uint(mloc));
}

__global__ void k_quant_qkv(const int* __restrict__ Y, signed char* __restrict__ out,
                            const float* __restrict__ dqf_p,
                            const unsigned* __restrict__ my_bits,
                            const float* __restrict__ clip, int transposeV) {
  const float dq = *dqf_p;
  const float s  = 127.0f / __uint_as_float(*my_bits);
  const float lo = clip[0], hic = clip[1];
  for (int i = blockIdx.x * blockDim.x + threadIdx.x; i < BSH_;
       i += gridDim.x * blockDim.x) {
    const int m = i / H_, n = i - m * H_;
    float y = fminf(fmaxf((float)Y[i] * dq, lo), hic);
    const signed char qv = (signed char)rintf(y * s);
    const int b = m >> 9, ss = m & 511;
    const int h = n >> 6, d = n & 63;
    const int idx = transposeV ? (((b * NH_ + h) * DH_ + d) * S_ + ss)
                               : (((b * NH_ + h) * S_ + ss) * DH_ + d);
    out[idx] = qv;
  }
}

__device__ __forceinline__ void compute_scores_lds(
    const signed char* __restrict__ Qq, const signed char* __restrict__ Kq,
    const float* __restrict__ mask, float* __restrict__ sc,
    int b, int h, int qb, float sscale) {
  const int lane = threadIdx.x & 31;
  const int wave = threadIdx.x >> 5;
  const int hi   = lane >> 4;
  const int l15  = lane & 15;

  const signed char* Qbase = Qq + ((b * NH_ + h) * S_ + qb * 16) * DH_;
  const signed char* Kbase = Kq + ((b * NH_ + h) * S_) * DH_;

  const v8i a = load_a_frag((const int*)(Qbase + l15 * DH_), 0, hi);

  v8i bb = load_b_frag((const int*)(Kbase + ((wave * 8) * 16 + l15) * DH_), 0, hi);
  for (int t = 0; t < 8; ++t) {
    const int kt = wave * 8 + t;
    v8i bbn = bb;
    if (t < 7)
      bbn = load_b_frag((const int*)(Kbase + ((kt + 1) * 16 + l15) * DH_), 0, hi);
    v8i c = {};
    c = __builtin_amdgcn_wmma_i32_16x16x64_iu8(true, a, true, bb, c,
                                               false, false);
#pragma unroll
    for (int r = 0; r < 8; ++r) {
      const int M = r + hi * 8;
      const int N = kt * 16 + l15;
      sc[M * S_ + N] = (float)c[r] * sscale + mask[b * S_ + N];
    }
    bb = bbn;
  }
}

__global__ void __launch_bounds__(128)
k_attn_phase1(const signed char* __restrict__ Qq, const signed char* __restrict__ Kq,
              const float* __restrict__ mask, const unsigned* __restrict__ mq_bits,
              const unsigned* __restrict__ mk_bits, unsigned* __restrict__ mp_bits,
              const float* __restrict__ aclip) {
  __shared__ float sc[16 * S_];
  const int blk = blockIdx.x;
  const int qb = blk & 31, bh = blk >> 5;
  const int h = bh % NH_, b = bh / NH_;
  const float sq = 127.0f / __uint_as_float(*mq_bits);
  const float sk = 127.0f / __uint_as_float(*mk_bits);
  compute_scores_lds(Qq, Kq, mask, sc, b, h, qb, 1.0f / (sq * sk * 8.0f));
  __syncthreads();

  const int lane = threadIdx.x & 31, wave = threadIdx.x >> 5;
  float mloc = 0.0f;
  for (int rr = 0; rr < 4; ++rr) {
    const int row = wave * 4 + rr;
    float mx = -3.4e38f;
    for (int c = lane; c < S_; c += 32) mx = fmaxf(mx, sc[row * S_ + c]);
    mx = wave_max_f(mx);
    float sum = 0.0f;
    for (int c = lane; c < S_; c += 32) sum += expf(sc[row * S_ + c] - mx);
    sum = wave_sum_f(sum);
    mloc = fmaxf(mloc, fminf(1.0f / sum, aclip[1]));
  }
  if (lane == 0) atomicMax(mp_bits, __float_as_uint(mloc));
}

__global__ void __launch_bounds__(128)
k_attn_phase2(const signed char* __restrict__ Qq, const signed char* __restrict__ Kq,
              const signed char* __restrict__ Vt, const float* __restrict__ mask,
              const unsigned* __restrict__ mq_bits, const unsigned* __restrict__ mk_bits,
              const unsigned* __restrict__ mv_bits, const unsigned* __restrict__ mp_bits,
              const float* __restrict__ aclip, float* __restrict__ out) {
  __shared__ float sc[16 * S_];
  __shared__ __align__(16) signed char pq[16 * S_];
  const int blk = blockIdx.x;
  const int qb = blk & 31, bh = blk >> 5;
  const int h = bh % NH_, b = bh / NH_;
  const float sq = 127.0f / __uint_as_float(*mq_bits);
  const float sk = 127.0f / __uint_as_float(*mk_bits);
  compute_scores_lds(Qq, Kq, mask, sc, b, h, qb, 1.0f / (sq * sk * 8.0f));
  __syncthreads();

  const int lane = threadIdx.x & 31, wave = threadIdx.x >> 5;
  const int hi = lane >> 4, l15 = lane & 15;
  const float sp = 127.0f / __uint_as_float(*mp_bits);
  const float plo = aclip[0], phi = aclip[1];
  for (int rr = 0; rr < 4; ++rr) {
    const int row = wave * 4 + rr;
    float mx = -3.4e38f;
    for (int c = lane; c < S_; c += 32) mx = fmaxf(mx, sc[row * S_ + c]);
    mx = wave_max_f(mx);
    float sum = 0.0f;
    for (int c = lane; c < S_; c += 32) sum += expf(sc[row * S_ + c] - mx);
    sum = wave_sum_f(sum);
    const float inv = 1.0f / sum;
    for (int c = lane; c < S_; c += 32) {
      float p = expf(sc[row * S_ + c] - mx) * inv;
      p = fminf(fmaxf(p, plo), phi);
      pq[row * S_ + c] = (signed char)rintf(p * sp);
    }
  }
  __syncthreads();

  const int nt = wave;
  const int* Pr = (const int*)(pq + l15 * S_);
  const int* Vr = (const int*)(Vt + ((b * NH_ + h) * DH_ + nt * 16 + l15) * S_);
  v8i acc = {};
  v8i a  = load_a_frag(Pr, 0, hi);
  v8i bb = load_b_frag(Vr, 0, hi);
  for (int kc = 64; kc < S_; kc += 64) {
    const int kd = kc >> 2;
    v8i an  = load_a_frag(Pr, kd, hi);
    v8i bbn = load_b_frag(Vr, kd, hi);
    acc = __builtin_amdgcn_wmma_i32_16x16x64_iu8(true, a, true, bb, acc,
                                                 false, false);
    a = an; bb = bbn;
  }
  acc = __builtin_amdgcn_wmma_i32_16x16x64_iu8(true, a, true, bb, acc,
                                               false, false);

  const float cs = 1.0f / (sp * (127.0f / __uint_as_float(*mv_bits)));
#pragma unroll
  for (int r = 0; r < 8; ++r) {
    const int srow = qb * 16 + r + hi * 8;
    out[(b * S_ + srow) * H_ + h * DH_ + nt * 16 + l15] = (float)acc[r] * cs;
  }
}

extern "C" void kernel_launch(void* const* d_in, const int* in_sizes, int n_in,
                              void* d_out, int out_size, void* d_ws, size_t ws_size,
                              hipStream_t stream) {
  const float* hs[2]  = {(const float*)d_in[0], (const float*)d_in[1]};
  const float* msk[2] = {(const float*)d_in[2], (const float*)d_in[3]};
  const float* W[2][3] = {
      {(const float*)d_in[4], (const float*)d_in[6], (const float*)d_in[8]},
      {(const float*)d_in[5], (const float*)d_in[7], (const float*)d_in[9]}};
  const float* inclip[2][3] = {
      {(const float*)d_in[10], (const float*)d_in[12], (const float*)d_in[14]},
      {(const float*)d_in[11], (const float*)d_in[13], (const float*)d_in[15]}};
  const float* yclip[2][3] = {
      {(const float*)d_in[16], (const float*)d_in[18], (const float*)d_in[20]},
      {(const float*)d_in[17], (const float*)d_in[19], (const float*)d_in[21]}};
  const float* aclip[2] = {(const float*)d_in[22], (const float*)d_in[23]};

  char* ws = (char*)d_ws;
  unsigned long long* wsum = (unsigned long long*)ws;
  unsigned* maxb = (unsigned*)(ws + 48);
  float* dqf = (float*)(ws + 128);
  signed char* wsign = (signed char*)(ws + 256);
  signed char* xq = wsign + 6 * (size_t)HH_;
  signed char* qq = xq + 2 * (size_t)BSH_;
  signed char* kq = qq + 2 * (size_t)BSH_;
  signed char* vt = kq + 2 * (size_t)BSH_;
  int* yi = (int*)(vt + 2 * (size_t)BSH_);

  k_init<<<1, 64, 0, stream>>>(wsum, maxb);

  for (int br = 0; br < 2; ++br)
    k_absmax_clip<<<512, 256, 0, stream>>>(hs[br], BSH_, inclip[br][0], maxb + br);

  for (int br = 0; br < 2; ++br)
    for (int p = 0; p < 3; ++p) {
      const int wi = br * 3 + p;
      k_wsign<<<256, 256, 0, stream>>>(W[br][p], wsign + (size_t)wi * HH_, wsum + wi);
    }

  k_scales<<<1, 64, 0, stream>>>(wsum, maxb, dqf);

  for (int br = 0; br < 2; ++br)
    k_quant_act<<<512, 256, 0, stream>>>(hs[br], xq + (size_t)br * BSH_, BSH_,
                                         inclip[br][0], maxb + br);

  for (int br = 0; br < 2; ++br)
    for (int p = 0; p < 3; ++p) {
      const int wi = br * 3 + p;
      signed char* dst = (p == 0 ? qq : (p == 1 ? kq : vt)) + (size_t)br * BSH_;
      k_proj_gemm<<<384, 256, 0, stream>>>(xq + (size_t)br * BSH_,
                                           wsign + (size_t)wi * HH_, yi,
                                           dqf + wi, yclip[br][p],
                                           maxb + 2 + wi);
      k_quant_qkv<<<512, 256, 0, stream>>>(yi, dst, dqf + wi, maxb + 2 + wi,
                                           yclip[br][p], p == 2 ? 1 : 0);
    }

  for (int br = 0; br < 2; ++br) {
    const size_t off = (size_t)br * BSH_;
    k_attn_phase1<<<3072, 128, 0, stream>>>(qq + off, kq + off, msk[br],
                                            maxb + 2 + br * 3 + 0,
                                            maxb + 2 + br * 3 + 1,
                                            maxb + 8 + br, aclip[br]);
    k_attn_phase2<<<3072, 128, 0, stream>>>(qq + off, kq + off, vt + off, msk[br],
                                            maxb + 2 + br * 3 + 0,
                                            maxb + 2 + br * 3 + 1,
                                            maxb + 2 + br * 3 + 2,
                                            maxb + 8 + br, aclip[br],
                                            (float*)d_out + off);
  }
}
